// UnslothMoEWrapper_5196910428756
// MI455X (gfx1250) — compile-verified, matrix operand order checked by reading, not run on hardware
//
#include <hip/hip_runtime.h>
#include <hip/hip_bf16.h>
#include <cstdint>
#include <math.h>

// ---------------------------------------------------------------------------
// MoE (top-4 of 32 experts), fp32 in/out, bf16 WMMA compute, TDM-fed LDS.
// Memory-bound on streaming 1.6 GB of fp32 expert weights -> grouped GEMM,
// weights DMA'd to LDS by the Tensor Data Mover with double buffering.
// ---------------------------------------------------------------------------

typedef __bf16 bf16;
typedef __attribute__((ext_vector_type(4)))  __bf16 v4bf;
typedef __attribute__((ext_vector_type(8)))  __bf16 v8bf;
typedef __attribute__((ext_vector_type(16))) __bf16 v16bf;
typedef __attribute__((ext_vector_type(4)))  float  v4f;
typedef __attribute__((ext_vector_type(8)))  float  v8f;
typedef __attribute__((ext_vector_type(4)))  unsigned int u32x4;
typedef __attribute__((ext_vector_type(4)))  int    i32x4;
typedef __attribute__((ext_vector_type(8)))  int    i32x8;

constexpr int Bc = 4, Sc = 2048, Hc = 2048, Ec = 32, Ic = 2048, TOPK = 4;
constexpr int Tc = Bc * Sc;                 // 8192 tokens
constexpr int TILE_M = 128, TILE_N = 64, TILE_K = 32;
constexpr int MAX_ROW_TILES = Tc / TILE_M;  // 64
constexpr int PADMAX = Tc * TOPK + Ec * TILE_M;  // 36864 padded pair slots

// ---- workspace layout (bytes), total ~185 MB ----
constexpr size_t WS_COUNTS  = 0;
constexpr size_t WS_CURSORS = 128;
constexpr size_t WS_OFFSETS = 256;
constexpr size_t WS_PAIRTOK = 512;
constexpr size_t WS_PAIRSCL = WS_PAIRTOK + (size_t)PADMAX * 4;
constexpr size_t WS_XBF     = (WS_PAIRSCL + (size_t)PADMAX * 4 + 255) & ~(size_t)255;
constexpr size_t WS_INTER   = WS_XBF + (size_t)Tc * Hc * 2;

// ---------------------------------------------------------------------------
// Tensor Data Mover: 2-D tile (tile_w elems x tile_h rows, row stride in elems)
// from global to LDS.  D# per CDNA5 ISA ch.8: group0 = count|lds|global|type,
// group1 = data_size | tensor dims | tile dims | dim0 stride.
// data_size_code: 0=1B,1=2B,2=4B,3=8B
__device__ __forceinline__ void tdm_load_2d(void* lds_ptr, const void* gptr,
                                            unsigned tile_w, unsigned tile_h,
                                            unsigned long long stride_elems,
                                            unsigned tensor_w,
                                            unsigned data_size_code) {
  unsigned long long ga = (unsigned long long)gptr;
  unsigned lds_off = (unsigned)(unsigned long long)lds_ptr;  // low 32b = LDS byte addr
  u32x4 g0;
  g0.x = 1u;                                            // count=1, user mode
  g0.y = lds_off;                                       // lds_addr
  g0.z = (unsigned)ga;                                  // global_addr[31:0]
  g0.w = (unsigned)((ga >> 32) & 0x1FFFFFFull) | (2u << 30);  // addr[56:32] | type=2
  unsigned td0 = tensor_w;                              // tensor_dim0 (OOB bound)
  unsigned td1 = 1u << 20;                              // tensor_dim1: plenty of rows
  unsigned long long st0 = stride_elems;                // tensor_dim0_stride (48b)
  i32x8 g1;
  g1[0] = (int)(data_size_code << 16);                  // wg_mask=0 | data_size
  g1[1] = (int)((td0 & 0xFFFFu) << 16);                 // atomic_bar=0 | td0.lo
  g1[2] = (int)((td0 >> 16) | ((td1 & 0xFFFFu) << 16)); // td0.hi | td1.lo
  g1[3] = (int)((td1 >> 16) | (tile_w << 16));          // td1.hi | tile_dim0
  g1[4] = (int)tile_h;                                  // tile_dim1 | tile_dim2=0
  g1[5] = (int)(st0 & 0xFFFFFFFFull);
  g1[6] = (int)((st0 >> 32) & 0xFFFFull);               // st0.hi | st1.lo=0
  g1[7] = 0;
  i32x4 gz = {0, 0, 0, 0};                              // 2-D tensor: groups 2/3 unused
#if __has_builtin(__builtin_amdgcn_tensor_load_to_lds)
#if defined(__clang_major__) && __clang_major__ >= 23
  i32x8 gz8 = {0, 0, 0, 0, 0, 0, 0, 0};
  __builtin_amdgcn_tensor_load_to_lds(g0, g1, gz, gz, gz8, 0);
#else
  __builtin_amdgcn_tensor_load_to_lds(g0, g1, gz, gz, 0);
#endif
#endif
}

__device__ __forceinline__ void tdm_wait() {
#if __has_builtin(__builtin_amdgcn_s_wait_tensorcnt)
  __builtin_amdgcn_s_wait_tensorcnt(0);                 // s_wait_tensorcnt 0
#endif
}

// ---------------------------------------------------------------------------
__global__ __launch_bounds__(256) void moe_cvt_bf16(const float* __restrict__ x,
                                                    bf16* __restrict__ y, int n) {
  int i = (blockIdx.x * blockDim.x + threadIdx.x) * 4;
  if (i >= n) return;
  float4 v = *(const float4*)(x + i);
  v4bf o = { (bf16)v.x, (bf16)v.y, (bf16)v.z, (bf16)v.w };
  *(v4bf*)(y + i) = o;
}

// ---------------------------------------------------------------------------
__device__ inline void top4(const float* __restrict__ l, int* bi, float* bv) {
  unsigned chosen = 0;
  for (int k = 0; k < TOPK; k++) {
    float best = -INFINITY; int besti = 0;
    for (int e = 0; e < Ec; e++) {
      if (chosen & (1u << e)) continue;
      float v = l[e];
      if (v > best) { best = v; besti = e; }
    }
    chosen |= 1u << besti; bi[k] = besti; bv[k] = best;
  }
}

__global__ __launch_bounds__(256) void moe_route_count(const float* __restrict__ logits,
                                                       int* __restrict__ counts) {
  int t = blockIdx.x * blockDim.x + threadIdx.x;
  if (t >= Tc) return;
  int bi[TOPK]; float bv[TOPK];
  top4(logits + (size_t)t * Ec, bi, bv);
  for (int k = 0; k < TOPK; k++) atomicAdd(&counts[bi[k]], 1);
}

__global__ void moe_scan(const int* __restrict__ counts, int* __restrict__ offsets) {
  if (threadIdx.x == 0 && blockIdx.x == 0) {
    int acc = 0;
    for (int e = 0; e < Ec; e++) {
      offsets[e] = acc;
      acc += (counts[e] + TILE_M - 1) & ~(TILE_M - 1);
    }
    offsets[Ec] = acc;
  }
}

__global__ __launch_bounds__(256) void moe_route_fill(const float* __restrict__ logits,
                                                      const int* __restrict__ offsets,
                                                      int* __restrict__ cursors,
                                                      int* __restrict__ pair_token,
                                                      float* __restrict__ pair_scale) {
  int t = blockIdx.x * blockDim.x + threadIdx.x;
  if (t >= Tc) return;
  int bi[TOPK]; float bv[TOPK];
  top4(logits + (size_t)t * Ec, bi, bv);
  float m = bv[0], w[TOPK], sum = 0.f;
  for (int k = 0; k < TOPK; k++) { w[k] = __expf(bv[k] - m); sum += w[k]; }
  float inv = 1.0f / sum;
  for (int k = 0; k < TOPK; k++) {
    int e = bi[k];
    int slot = atomicAdd(&cursors[e], 1);
    int p = offsets[e] + slot;
    pair_token[p] = t;
    pair_scale[p] = w[k] * inv;
  }
}

// ---------------------------------------------------------------------------
// Fragment gathers per ISA 7.12.2: lanes 0-15 -> K {0..7,16..23};
// lanes 16-31 -> K {8..15,24..31}.  Row-major 32-wide LDS slabs.
__device__ __forceinline__ v16bf load_frag(const bf16* __restrict__ rowp, int kg) {
  v8bf lo = *(const v8bf*)(rowp + kg * 8);
  v8bf hi = *(const v8bf*)(rowp + 16 + kg * 8);
  return __builtin_shufflevector(lo, hi, 0,1,2,3,4,5,6,7,8,9,10,11,12,13,14,15);
}

// Same gather from an fp32 LDS slab (TDM copies weights raw), cvt -> bf16.
__device__ __forceinline__ v16bf load_frag_f32(const float* __restrict__ rowp, int kg) {
  const v4f* q = (const v4f*)(rowp + kg * 8);
  const v4f* r = (const v4f*)(rowp + 16 + kg * 8);
  v4f a0 = q[0], a1 = q[1], b0 = r[0], b1 = r[1];
  v16bf f = { (bf16)a0.x,(bf16)a0.y,(bf16)a0.z,(bf16)a0.w,
              (bf16)a1.x,(bf16)a1.y,(bf16)a1.z,(bf16)a1.w,
              (bf16)b0.x,(bf16)b0.y,(bf16)b0.z,(bf16)b0.w,
              (bf16)b1.x,(bf16)b1.y,(bf16)b1.z,(bf16)b1.w };
  return f;
}

// ---------------------------------------------------------------------------
// GEMM1: inter[pair, i] = silu(x @ w13_gate^T) * (x @ w13_up^T), grouped by
// expert.  grid = (I/64, MAX_ROW_TILES, E); 256 threads = 8 wave32 (4M x 2N).
// Weight tiles are DMA'd into LDS (fp32) by the TDM, double buffered; the X
// tile is a token-gather done with per-thread vector loads into LDS (bf16).
__global__ __launch_bounds__(256)
void moe_gemm1(const bf16* __restrict__ xbf, const float* __restrict__ w13,
               const int* __restrict__ pair_token, const int* __restrict__ offsets,
               bf16* __restrict__ inter) {
  const int e   = blockIdx.z;
  const int off = offsets[e];
  const int pc  = offsets[e + 1] - off;
  if ((int)blockIdx.y * TILE_M >= pc) return;
  const int row0 = off + blockIdx.y * TILE_M;
  const int nb   = blockIdx.x * TILE_N;

  __shared__ int   ltok[TILE_M];
  __shared__ bf16  lX [2][TILE_M * TILE_K];   // 2 x 8 KB
  __shared__ float lWg[2][TILE_N * TILE_K];   // 2 x 8 KB (fp32 via TDM)
  __shared__ float lWu[2][TILE_N * TILE_K];   // 2 x 8 KB

  const int tid  = threadIdx.x;
  const int lane = tid & 31;
  const int wid  = tid >> 5;
  const int wm   = wid & 3;
  const int wn   = wid >> 2;
  const int kg   = lane >> 4;
  const int lr   = lane & 15;

  if (tid < TILE_M) {
    int tk = pair_token[row0 + tid];
    ltok[tid] = tk < 0 ? 0 : tk;   // padded rows read token 0; outputs discarded later
  }
  __syncthreads();

  const int xr = tid >> 1;               // X row this thread loads
  const int xh = (tid & 1) * 16;         // 16 bf16 (32 B) half-row

  const float* wg_base = w13 + ((size_t)e * (2 * Ic) + nb) * Hc;
  const float* wu_base = w13 + ((size_t)e * (2 * Ic) + Ic + nb) * Hc;

  // ---- stage loader: TDM for weights (wave 0), vector gather for X ----
  auto stage = [&](int kk, int buf) {
    if (wid == 0) {
      tdm_load_2d(&lWg[buf][0], wg_base + kk, TILE_K, TILE_N, Hc, Hc, 2u);
      tdm_load_2d(&lWu[buf][0], wu_base + kk, TILE_K, TILE_N, Hc, Hc, 2u);
    }
    const uint4* src = (const uint4*)(xbf + (size_t)ltok[xr] * Hc + kk + xh);
    uint4* dst = (uint4*)(&lX[buf][0] + xr * TILE_K + xh);
    dst[0] = src[0];
    dst[1] = src[1];
  };

  v8f zero = {0.f,0.f,0.f,0.f,0.f,0.f,0.f,0.f};
  v8f accG[2][2], accU[2][2];
  for (int i = 0; i < 2; i++)
    for (int j = 0; j < 2; j++) { accG[i][j] = zero; accU[i][j] = zero; }

  // prologue: fill buffer 0
  stage(0, 0);
  tdm_wait();
  __syncthreads();

  constexpr int NK = Hc / TILE_K;
  for (int k = 0; k < NK; k++) {
    const int cur = k & 1;
    if (k + 1 < NK) stage((k + 1) * TILE_K, cur ^ 1);   // prefetch next slab

    v16bf a[2];
    for (int sm = 0; sm < 2; sm++)
      a[sm] = load_frag(&lX[cur][0] + (wm * 32 + sm * 16 + lr) * TILE_K, kg);
    for (int sn = 0; sn < 2; sn++) {
      v16bf bg = load_frag_f32(&lWg[cur][0] + (wn * 32 + sn * 16 + lr) * TILE_K, kg);
      v16bf bu = load_frag_f32(&lWu[cur][0] + (wn * 32 + sn * 16 + lr) * TILE_K, kg);
      for (int sm = 0; sm < 2; sm++) {
        accG[sm][sn] = __builtin_amdgcn_wmma_f32_16x16x32_bf16(
            false, a[sm], false, bg, (short)0, accG[sm][sn], false, false);
        accU[sm][sn] = __builtin_amdgcn_wmma_f32_16x16x32_bf16(
            false, a[sm], false, bu, (short)0, accU[sm][sn], false, false);
      }
    }
    tdm_wait();          // prefetch into buf^1 complete (no-op for waves 1..7)
    __syncthreads();     // everyone done reading cur, buf^1 fully staged
  }

  // D layout: VGPR r -> M = r (lanes 0-15) / 8+r (lanes 16-31); N = lane&15
  const int mofs = (lane < 16) ? 0 : 8;
  for (int sm = 0; sm < 2; sm++)
    for (int sn = 0; sn < 2; sn++) {
      int icol = nb + wn * 32 + sn * 16 + lr;
      for (int r = 0; r < 8; r++) {
        float g = accG[sm][sn][r];
        float u = accU[sm][sn][r];
        float s = (g / (1.0f + __expf(-g))) * u;       // silu(g) * u
        int prow = row0 + wm * 32 + sm * 16 + mofs + r;
        inter[(size_t)prow * Ic + icol] = (bf16)s;
      }
    }
}

// ---------------------------------------------------------------------------
// GEMM2: out[token, h] += scale * (inter[pair, :] @ w2_e[h, :]^T)
// Both tiles are contiguous 2-D -> both DMA'd by the TDM, double buffered.
__global__ __launch_bounds__(256)
void moe_gemm2(const bf16* __restrict__ inter, const float* __restrict__ w2,
               const int* __restrict__ pair_token, const float* __restrict__ pair_scale,
               const int* __restrict__ offsets, float* __restrict__ out) {
  const int e   = blockIdx.z;
  const int off = offsets[e];
  const int pc  = offsets[e + 1] - off;
  if ((int)blockIdx.y * TILE_M >= pc) return;
  const int row0 = off + blockIdx.y * TILE_M;
  const int hb   = blockIdx.x * TILE_N;

  __shared__ int   ltok[TILE_M];
  __shared__ float lscl[TILE_M];
  __shared__ bf16  lA[2][TILE_M * TILE_K];    // 2 x 8 KB (bf16 via TDM)
  __shared__ float lW[2][TILE_N * TILE_K];    // 2 x 8 KB (fp32 via TDM)

  const int tid  = threadIdx.x;
  const int lane = tid & 31;
  const int wid  = tid >> 5;
  const int wm   = wid & 3;
  const int wn   = wid >> 2;
  const int kg   = lane >> 4;
  const int lr   = lane & 15;

  if (tid < TILE_M) {
    ltok[tid] = pair_token[row0 + tid];
    lscl[tid] = pair_scale[row0 + tid];
  }

  const bf16*  a_base = inter + (size_t)row0 * Ic;
  const float* w_base = w2 + ((size_t)e * Hc + hb) * Ic;

  auto stage = [&](int kk, int buf) {
    if (wid == 0) {
      tdm_load_2d(&lA[buf][0], a_base + kk, TILE_K, TILE_M, Ic, Ic, 1u);  // bf16
      tdm_load_2d(&lW[buf][0], w_base + kk, TILE_K, TILE_N, Ic, Ic, 2u);  // fp32
    }
  };

  v8f zero = {0.f,0.f,0.f,0.f,0.f,0.f,0.f,0.f};
  v8f acc[2][2];
  for (int i = 0; i < 2; i++) for (int j = 0; j < 2; j++) acc[i][j] = zero;

  stage(0, 0);
  tdm_wait();
  __syncthreads();

  constexpr int NK = Ic / TILE_K;
  for (int k = 0; k < NK; k++) {
    const int cur = k & 1;
    if (k + 1 < NK) stage((k + 1) * TILE_K, cur ^ 1);

    v16bf a[2];
    for (int sm = 0; sm < 2; sm++)
      a[sm] = load_frag(&lA[cur][0] + (wm * 32 + sm * 16 + lr) * TILE_K, kg);
    for (int sn = 0; sn < 2; sn++) {
      v16bf b = load_frag_f32(&lW[cur][0] + (wn * 32 + sn * 16 + lr) * TILE_K, kg);
      for (int sm = 0; sm < 2; sm++)
        acc[sm][sn] = __builtin_amdgcn_wmma_f32_16x16x32_bf16(
            false, a[sm], false, b, (short)0, acc[sm][sn], false, false);
    }
    tdm_wait();
    __syncthreads();
  }

  const int mofs = (lane < 16) ? 0 : 8;
  for (int sm = 0; sm < 2; sm++)
    for (int sn = 0; sn < 2; sn++) {
      int hcol = hb + wn * 32 + sn * 16 + lr;
      for (int r = 0; r < 8; r++) {
        int lrow = wm * 32 + sm * 16 + mofs + r;
        int tok  = ltok[lrow];
        if (tok >= 0)
          atomicAdd(&out[(size_t)tok * Hc + hcol], lscl[lrow] * acc[sm][sn][r]);
      }
    }
}

// ---------------------------------------------------------------------------
extern "C" void kernel_launch(void* const* d_in, const int* in_sizes, int n_in,
                              void* d_out, int out_size, void* d_ws, size_t ws_size,
                              hipStream_t stream) {
  const float* hs  = (const float*)d_in[0];   // [B,S,H] fp32
  const float* rl  = (const float*)d_in[1];   // [B,S,E] fp32
  const float* w13 = (const float*)d_in[2];   // [E,2I,H] fp32
  const float* w2  = (const float*)d_in[3];   // [E,H,I] fp32
  float* out = (float*)d_out;

  char* ws = (char*)d_ws;
  int*   counts     = (int*)(ws + WS_COUNTS);
  int*   cursors    = (int*)(ws + WS_CURSORS);
  int*   offsets    = (int*)(ws + WS_OFFSETS);
  int*   pair_token = (int*)(ws + WS_PAIRTOK);
  float* pair_scale = (float*)(ws + WS_PAIRSCL);
  bf16*  xbf        = (bf16*)(ws + WS_XBF);
  bf16*  inter      = (bf16*)(ws + WS_INTER);

  hipMemsetAsync(ws, 0, 512, stream);                             // counts/cursors/offsets
  hipMemsetAsync(pair_token, 0xFF, (size_t)PADMAX * 4, stream);   // -1 padding
  hipMemsetAsync(out, 0, (size_t)out_size * sizeof(float), stream);

  moe_cvt_bf16<<<(Tc * Hc / 4 + 255) / 256, 256, 0, stream>>>(hs, xbf, Tc * Hc);
  moe_route_count<<<Tc / 256, 256, 0, stream>>>(rl, counts);
  moe_scan<<<1, 32, 0, stream>>>(counts, offsets);
  moe_route_fill<<<Tc / 256, 256, 0, stream>>>(rl, offsets, cursors, pair_token, pair_scale);

  moe_gemm1<<<dim3(Ic / TILE_N, MAX_ROW_TILES, Ec), 256, 0, stream>>>(
      xbf, w13, pair_token, offsets, inter);
  moe_gemm2<<<dim3(Hc / TILE_N, MAX_ROW_TILES, Ec), 256, 0, stream>>>(
      inter, w2, pair_token, pair_scale, offsets, out);
}
